// JangDFlashAttention_47983374631189
// MI455X (gfx1250) — matrix operand order repaired and checked by reading, not compiled, not run on hardware
//
#include <hip/hip_runtime.h>

// ---------------------------------------------------------------------------
// Model constants (match the JAX reference)
// ---------------------------------------------------------------------------
#define BATCH   2
#define SEQ_L   2048
#define SEQ_T   512
#define SEQ_S   (SEQ_T + SEQ_L)   // 2560 keys total
#define HID     2048
#define NHEADS  16
#define NKVH    4
#define HDIM    128
#define QOUT    (NHEADS * HDIM)   // 2048
#define KVOUT   (NKVH * HDIM)     // 512
#define NTOK    (BATCH * SEQ_L)   // 4096 block tokens
#define NCTX    (BATCH * SEQ_T)   // 1024 ctx tokens
#define SCALE_QK 0.08838834764831845f   // 1/sqrt(128), folded into Q

typedef __attribute__((ext_vector_type(16))) __bf16 v16bf;
typedef __attribute__((ext_vector_type(8)))  float  v8f;

// 32B fragment <-> two 128-bit loads
union F16 {
    v16bf v;
    uint4 q[2];
};

__device__ __forceinline__ v8f zero_v8f() {
    v8f z;
#pragma unroll
    for (int i = 0; i < 8; ++i) z[i] = 0.0f;
    return z;
}

// ---------------------------------------------------------------------------
// DPP16 butterfly reductions within each 16-lane row (no LDS traffic).
// quad_perm(1,0,3,2)=0xB1, quad_perm(2,3,0,1)=0x4E,
// row_half_mirror=0x141, row_mirror=0x140 — all stay inside a 16-lane row.
// dpp_ctrl must be an ICE -> template parameter.
// ---------------------------------------------------------------------------
template <int CTRL>
__device__ __forceinline__ float dpp_xchg(float x) {
    return __int_as_float(__builtin_amdgcn_update_dpp(
        0, __float_as_int(x), CTRL, 0xF, 0xF, true));
}
__device__ __forceinline__ float row_reduce_max16(float x) {
    x = fmaxf(x, dpp_xchg<0xB1>(x));
    x = fmaxf(x, dpp_xchg<0x4E>(x));
    x = fmaxf(x, dpp_xchg<0x141>(x));
    x = fmaxf(x, dpp_xchg<0x140>(x));
    return x;
}
__device__ __forceinline__ float row_reduce_sum16(float x) {
    x += dpp_xchg<0xB1>(x);
    x += dpp_xchg<0x4E>(x);
    x += dpp_xchg<0x141>(x);
    x += dpp_xchg<0x140>(x);
    return x;
}

// ---------------------------------------------------------------------------
// f32 -> bf16 conversion (grid-stride)
// ---------------------------------------------------------------------------
__global__ void cvt_f32_bf16(const float* __restrict__ in, __bf16* __restrict__ out,
                             size_t n) {
    size_t i = (size_t)blockIdx.x * blockDim.x + threadIdx.x;
    size_t stride = (size_t)gridDim.x * blockDim.x;
    for (; i < n; i += stride) out[i] = (__bf16)in[i];
}

// ---------------------------------------------------------------------------
// Generic bf16 WMMA GEMM: C[M,N] = A[M,K] @ B[K,N]
// A,B row-major bf16.  OUTF32=1 -> f32 out, else bf16 out.
// Block: 256 threads = 8 waves, tile 128(M) x 64(N), K-step 32.
// Waves arranged 4(M) x 2(N); each wave owns a 32x32 tile = 2x2 WMMA frags.
// LDS rows padded (40 / 72 bf16) for conflict-free b128 fragment reads.
// ---------------------------------------------------------------------------
template <int OUTF32>
__global__ __launch_bounds__(256) void gemm_bf16_wmma(
    const __bf16* __restrict__ A, const __bf16* __restrict__ B,
    void* __restrict__ Cout, int M, int N, int K) {
    const int m0 = blockIdx.y * 128;
    const int n0 = blockIdx.x * 64;
    const int tid  = threadIdx.x;
    const int wave = tid >> 5;
    const int lane = tid & 31;
    const int lh   = lane & 15;
    const bool hi  = lane >= 16;
    const int wm = wave >> 1;   // 0..3
    const int wn = wave & 1;    // 0..1

    __shared__ __align__(16) __bf16 As[128][40];   // 128x32 used, row pad -> 20 banks
    __shared__ __align__(16) __bf16 Bs[32][72];    // 32x64 used,  row pad -> 36 banks

    v8f acc[2][2];
#pragma unroll
    for (int mi = 0; mi < 2; ++mi)
#pragma unroll
        for (int ni = 0; ni < 2; ++ni) acc[mi][ni] = zero_v8f();

    for (int k0 = 0; k0 < K; k0 += 32) {
        __syncthreads();
        // ---- stage A tile (128x32 bf16): 512 16B-chunks, 2 per thread
#pragma unroll
        for (int i = 0; i < 2; ++i) {
            int c = tid + i * 256;          // 0..511
            int row = c >> 2, part = c & 3; // 4 x 8 bf16 per row
            *(uint4*)&As[row][part * 8] =
                *(const uint4*)&A[(size_t)(m0 + row) * K + k0 + part * 8];
        }
        // ---- stage B tile (32x64 bf16): 256 16B-chunks, 1 per thread
        {
            int row = tid >> 3, part = tid & 7;
            *(uint4*)&Bs[row][part * 8] =
                *(const uint4*)&B[(size_t)(k0 + row) * N + n0 + part * 8];
        }
        __syncthreads();

        // ---- A fragments (16x32): lane=M row; K 0-7/16-23 (lo) or 8-15/24-31 (hi)
        F16 af[2];
        const int ofs = hi ? 8 : 0;
#pragma unroll
        for (int mi = 0; mi < 2; ++mi) {
            int row = wm * 32 + mi * 16 + lh;
            af[mi].q[0] = *(const uint4*)&As[row][ofs];
            af[mi].q[1] = *(const uint4*)&As[row][ofs + 16];
        }
        // ---- B fragments (32x16): lane = N col; K 0-7/16-23 (lo) or 8-15/24-31 (hi)
        F16 bf2[2];
#pragma unroll
        for (int ni = 0; ni < 2; ++ni) {
            int col = wn * 32 + ni * 16;
#pragma unroll
            for (int s = 0; s < 16; ++s)
                bf2[ni].v[s] = Bs[(s < 8 ? ofs + s : 8 + ofs + s)][col + lh];
        }
#pragma unroll
        for (int mi = 0; mi < 2; ++mi)
#pragma unroll
            for (int ni = 0; ni < 2; ++ni)
                acc[mi][ni] = __builtin_amdgcn_wmma_f32_16x16x32_bf16(
                    false, af[mi].v, false, bf2[ni].v, (short)0, acc[mi][ni],
                    false, false);
    }

    // ---- store: C frag VGPR r -> row r (lanes 0-15) / r+8 (lanes 16-31)
#pragma unroll
    for (int mi = 0; mi < 2; ++mi)
#pragma unroll
        for (int ni = 0; ni < 2; ++ni)
#pragma unroll
            for (int r = 0; r < 8; ++r) {
                int row = m0 + wm * 32 + mi * 16 + r + (hi ? 8 : 0);
                int col = n0 + wn * 32 + ni * 16 + lh;
                if (OUTF32)
                    ((float*)Cout)[(size_t)row * N + col] = acc[mi][ni][r];
                else
                    ((__bf16*)Cout)[(size_t)row * N + col] = (__bf16)acc[mi][ni][r];
            }
}

// ---------------------------------------------------------------------------
// Per-(token, head) RMS-norm (+optional RoPE), f32 math, bf16 in/out.
// in : [B, ntok, nheads, 128]
// out (dmajor=0): [B, nheads, out_seq, 128]   at seq index seq_off+s
// out (dmajor=1): [B, nheads, 128, out_seq]   at seq index seq_off+s  (K cache)
// out_scale: folded softmax scale (1/sqrt(D) for Q, 1.0 otherwise).
// One block = one (b, s, h); 128 threads (one per d).
// ---------------------------------------------------------------------------
__global__ __launch_bounds__(128) void post_norm_rope(
    const __bf16* __restrict__ in, __bf16* __restrict__ out,
    const float* __restrict__ normw, int ntok, int nheads, int out_seq,
    int seq_off, int do_rope, int dmajor, float out_scale) {
    int gid = blockIdx.x;
    int h = gid % nheads;
    int s = (gid / nheads) % ntok;
    int b = gid / (nheads * ntok);
    int d = threadIdx.x;

    float val = (float)in[(((size_t)(b * ntok + s)) * nheads + h) * HDIM + d];

    __shared__ float red[4];
    __shared__ float ybuf[HDIM];
    float sq = val * val;
#pragma unroll
    for (int m = 16; m >= 1; m >>= 1) sq += __shfl_xor(sq, m, 32);
    if ((threadIdx.x & 31) == 0) red[threadIdx.x >> 5] = sq;
    __syncthreads();
    float var = (red[0] + red[1] + red[2] + red[3]) * (1.0f / 128.0f);
    float y = val * rsqrtf(var + 1e-6f) * normw[d];

    float res = y;
    if (do_rope) {
        ybuf[d] = y;
        __syncthreads();
        int i2 = d & 63;
        // inv_freq = theta^(-2*i2/128) = exp(-(i2/64)*ln(10000))
        float inv = __expf(-((float)i2 * (1.0f / 64.0f)) * 9.210340371976184f);
        float ang = (float)s * inv;
        float c  = __cosf(ang);
        float sn = __sinf(ang);
        float rot = (d < 64) ? -ybuf[d + 64] : ybuf[d - 64];
        res = y * c + rot * sn;
    }
    res *= out_scale;
    size_t oidx;
    if (dmajor)
        oidx = (((size_t)(b * nheads + h)) * HDIM + d) * out_seq + seq_off + s;
    else
        oidx = (((size_t)(b * nheads + h)) * out_seq + seq_off + s) * HDIM + d;
    out[oidx] = (__bf16)res;
}

// ---------------------------------------------------------------------------
// Scatter V: in [B, ntok, NKVH, 128] -> out [B, NKVH, SEQ_S, 128] at seq_off
// ---------------------------------------------------------------------------
__global__ void scatter_v(const __bf16* __restrict__ in,
                          __bf16* __restrict__ out, int ntok, int seq_off,
                          size_t n) {
    size_t i = (size_t)blockIdx.x * blockDim.x + threadIdx.x;
    size_t stride = (size_t)gridDim.x * blockDim.x;
    for (; i < n; i += stride) {
        int d = (int)(i % HDIM);
        int h = (int)((i / HDIM) % NKVH);
        int s = (int)((i / (HDIM * NKVH)) % ntok);
        int b = (int)(i / ((size_t)HDIM * NKVH * ntok));
        out[(((size_t)(b * NKVH + h)) * SEQ_S + seq_off + s) * HDIM + d] = in[i];
    }
}

// ---------------------------------------------------------------------------
// One 32-key attention tile (templated so the bulk loop carries no mask code).
// ---------------------------------------------------------------------------
template <bool MASKED>
__device__ __forceinline__ void attn_tile(
    int kb, int qrow, int lane, int lh, int ofs, int wave,
    const __bf16* __restrict__ kbase, const __bf16* __restrict__ vbase,
    const F16 (&qf)[4], v8f (&o)[8], float (&m_r)[8], float (&l_r)[8],
    __bf16 (*Ps)[40]) {
    // ---- S = Q @ K^T: B-frag lane = key col; d 0-7/16-23 (lo) or 8-15/24-31 (hi)
    v8f sA = zero_v8f(), sB = zero_v8f();
#pragma unroll
    for (int c = 0; c < 4; ++c) {
        const __bf16* kr = kbase + ((size_t)(c * 32)) * SEQ_S + kb + lh;
        F16 ka, kbf;
#pragma unroll
        for (int s = 0; s < 16; ++s) {
            size_t dk = (size_t)(s < 8 ? ofs + s : 8 + ofs + s) * SEQ_S;
            ka.v[s]  = kr[dk];
            kbf.v[s] = kr[dk + 16];
        }
        sA = __builtin_amdgcn_wmma_f32_16x16x32_bf16(false, qf[c].v, false,
                                                     ka.v, (short)0, sA,
                                                     false, false);
        sB = __builtin_amdgcn_wmma_f32_16x16x32_bf16(false, qf[c].v, false,
                                                     kbf.v, (short)0, sB,
                                                     false, false);
    }

    // ---- causal mask (diagonal tiles only; scale pre-folded into Q)
    if (MASKED) {
#pragma unroll
        for (int r = 0; r < 8; ++r) {
            int i  = qrow + r;
            int ja = kb + lh - SEQ_T;
            if (ja > i)      sA[r] = -1e30f;
            if (ja + 16 > i) sB[r] = -1e30f;
        }
    }

    // ---- online softmax: row r lives in one 16-lane row -> DPP16 butterfly
#pragma unroll
    for (int r = 0; r < 8; ++r) {
        float mx = row_reduce_max16(fmaxf(sA[r], sB[r]));
        float mn = fmaxf(m_r[r], mx);
        float alpha = __expf(m_r[r] - mn);
        float pa = __expf(sA[r] - mn);
        float pb = __expf(sB[r] - mn);
        sA[r] = pa; sB[r] = pb;
        float rs = row_reduce_sum16(pa + pb);
        l_r[r] = l_r[r] * alpha + rs;
        m_r[r] = mn;
#pragma unroll
        for (int n = 0; n < 8; ++n) o[n][r] *= alpha;
    }

    // ---- re-layout P (C-frag) -> A-frag via per-wave LDS scratch
#pragma unroll
    for (int r = 0; r < 8; ++r) {
        Ps[ofs + r][lh]      = (__bf16)sA[r];
        Ps[ofs + r][16 + lh] = (__bf16)sB[r];
    }
    asm volatile("s_wait_dscnt 0x0" ::: "memory");
    F16 pf;
    pf.q[0] = *(const uint4*)&Ps[lh][ofs];
    pf.q[1] = *(const uint4*)&Ps[lh][ofs + 16];

    // ---- O += P @ V : B-frag lane = d col; keys 0-7/16-23 (lo) or 8-15/24-31 (hi)
#pragma unroll
    for (int n = 0; n < 8; ++n) {
        const __bf16* vr = vbase + ((size_t)kb) * HDIM + n * 16 + lh;
        F16 vf;
#pragma unroll
        for (int s = 0; s < 16; ++s)
            vf.v[s] = vr[(size_t)(s < 8 ? ofs + s : 8 + ofs + s) * HDIM];
        o[n] = __builtin_amdgcn_wmma_f32_16x16x32_bf16(
            false, pf.v, false, vf.v, (short)0, o[n], false, false);
    }
}

// ---------------------------------------------------------------------------
// Flash attention (bf16 WMMA, f32 online softmax), no K/V LDS staging.
// Q:[B,H,L,128] (pre-scaled by 1/sqrt(D))  K:[B,NKVH,128,SEQ_S] (d-major)
// V:[B,NKVH,SEQ_S,128]   O:[B*L, H*128] (bf16)
// grid = (L/64, B*H), block = 128 (4 independent waves x 16 query rows).
// K/V fragments are gathered by 16 strided bf16 loads each; KV caches (~10MB) stay in L2.
// Bulk key loop is mask-free; exactly two diagonal tiles take the masked path.
// ---------------------------------------------------------------------------
__global__ __launch_bounds__(128) void attn_wmma(
    const __bf16* __restrict__ Q, const __bf16* __restrict__ Kf,
    const __bf16* __restrict__ Vf, __bf16* __restrict__ O) {
    const int bh = blockIdx.y;
    const int h = bh % NHEADS;
    const int b = bh / NHEADS;
    const int kvh = h >> 2;            // GQA: 4 q-heads per kv-head
    const int q0 = blockIdx.x * 64;
    const int wave = threadIdx.x >> 5;
    const int lane = threadIdx.x & 31;
    const int lh = lane & 15;
    const bool hi = lane >= 16;
    const int ofs = hi ? 8 : 0;

    // per-wave P scratch for C-frag -> A-frag relayout (row pad -> 20 banks)
    __shared__ __align__(16) __bf16 Ps[4][16][40];

    // ---- Q A-fragments for this wave's 16 rows (4 chunks of K=32)
    const __bf16* qbase =
        Q + (((size_t)bh * SEQ_L) + q0 + wave * 16 + lh) * HDIM;
    F16 qf[4];
#pragma unroll
    for (int c = 0; c < 4; ++c) {
        qf[c].q[0] = *(const uint4*)(qbase + c * 32 + ofs);
        qf[c].q[1] = *(const uint4*)(qbase + c * 32 + ofs + 16);
    }

    v8f o[8];
#pragma unroll
    for (int n = 0; n < 8; ++n) o[n] = zero_v8f();
    float m_r[8], l_r[8];
#pragma unroll
    for (int r = 0; r < 8; ++r) { m_r[r] = -1e30f; l_r[r] = 0.0f; }

    const __bf16* kbase = Kf + ((size_t)(b * NKVH + kvh)) * HDIM * SEQ_S;
    const __bf16* vbase = Vf + ((size_t)(b * NKVH + kvh)) * SEQ_S * HDIM;
    const int qrow = q0 + wave * 16 + (hi ? 8 : 0);

    // ---- bulk tiles: ctx keys + strictly-below-diagonal block keys (no mask)
    const int kdiag = SEQ_T + q0;
    for (int kb = 0; kb < kdiag; kb += 32)
        attn_tile<false>(kb, qrow, lane, lh, ofs, wave, kbase, vbase, qf, o,
                         m_r, l_r, Ps[wave]);
    // ---- two diagonal tiles (64 queries x 2x32 keys), masked path
    attn_tile<true>(kdiag, qrow, lane, lh, ofs, wave, kbase, vbase, qf, o,
                    m_r, l_r, Ps[wave]);
    attn_tile<true>(kdiag + 32, qrow, lane, lh, ofs, wave, kbase, vbase, qf, o,
                    m_r, l_r, Ps[wave]);

    // ---- epilogue: O /= l, write [B*L, H*128] bf16
#pragma unroll
    for (int r = 0; r < 8; ++r) {
        float inv = 1.0f / l_r[r];
        int row = q0 + wave * 16 + r + (hi ? 8 : 0);
        size_t base = ((size_t)(b * SEQ_L + row)) * QOUT + h * HDIM + lh;
#pragma unroll
        for (int n = 0; n < 8; ++n)
            O[base + n * 16] = (__bf16)(o[n][r] * inv);
    }
}

// ---------------------------------------------------------------------------
// Host orchestration
// ---------------------------------------------------------------------------
extern "C" void kernel_launch(void* const* d_in, const int* in_sizes, int n_in,
                              void* d_out, int out_size, void* d_ws,
                              size_t ws_size, hipStream_t stream) {
    (void)in_sizes; (void)n_in; (void)out_size; (void)ws_size;

    const float* x    = (const float*)d_in[0];
    const float* hctx = (const float*)d_in[1];
    const float* wq   = (const float*)d_in[2];
    const float* wk   = (const float*)d_in[3];
    const float* wv   = (const float*)d_in[4];
    const float* wo   = (const float*)d_in[5];
    const float* wkc  = (const float*)d_in[6];
    const float* wvc  = (const float*)d_in[7];
    const float* qnw  = (const float*)d_in[8];
    const float* knw  = (const float*)d_in[9];
    float* out = (float*)d_out;

    // ---- workspace carve-out (bf16 buffers, 256B aligned)
    char* wp = (char*)d_ws;
    auto alloc = [&](size_t elems) {
        char* p = wp;
        wp += (elems * sizeof(__bf16) + 255) & ~(size_t)255;
        return (__bf16*)p;
    };
    __bf16* x_bf    = alloc((size_t)NTOK * HID);
    __bf16* hctx_bf = alloc((size_t)NCTX * HID);
    __bf16* wq_bf   = alloc((size_t)HID * QOUT);
    __bf16* wk_bf   = alloc((size_t)HID * KVOUT);
    __bf16* wv_bf   = alloc((size_t)HID * KVOUT);
    __bf16* wo_bf   = alloc((size_t)QOUT * HID);
    __bf16* wkc_bf  = alloc((size_t)HID * KVOUT);
    __bf16* wvc_bf  = alloc((size_t)HID * KVOUT);
    __bf16* q_raw   = alloc((size_t)NTOK * QOUT);
    __bf16* k_raw   = alloc((size_t)NTOK * KVOUT);
    __bf16* v_raw   = alloc((size_t)NTOK * KVOUT);
    __bf16* kc_raw  = alloc((size_t)NCTX * KVOUT);
    __bf16* vc_raw  = alloc((size_t)NCTX * KVOUT);
    __bf16* q_att   = alloc((size_t)BATCH * NHEADS * SEQ_L * HDIM);
    __bf16* k_full  = alloc((size_t)BATCH * NKVH * HDIM * SEQ_S);  // d-major
    __bf16* v_full  = alloc((size_t)BATCH * NKVH * SEQ_S * HDIM);
    __bf16* attn_bf = alloc((size_t)NTOK * QOUT);

    // ---- 1) convert everything to bf16
    auto cvt = [&](const float* src, __bf16* dst, size_t n) {
        cvt_f32_bf16<<<2048, 256, 0, stream>>>(src, dst, n);
    };
    cvt(x,    x_bf,    (size_t)NTOK * HID);
    cvt(hctx, hctx_bf, (size_t)NCTX * HID);
    cvt(wq,   wq_bf,   (size_t)HID * QOUT);
    cvt(wk,   wk_bf,   (size_t)HID * KVOUT);
    cvt(wv,   wv_bf,   (size_t)HID * KVOUT);
    cvt(wo,   wo_bf,   (size_t)QOUT * HID);
    cvt(wkc,  wkc_bf,  (size_t)HID * KVOUT);
    cvt(wvc,  wvc_bf,  (size_t)HID * KVOUT);

    // ---- 2) projection GEMMs (bf16 in, f32 accum, bf16 out)
    gemm_bf16_wmma<0><<<dim3(QOUT / 64, NTOK / 128), 256, 0, stream>>>(
        x_bf, wq_bf, q_raw, NTOK, QOUT, HID);
    gemm_bf16_wmma<0><<<dim3(KVOUT / 64, NTOK / 128), 256, 0, stream>>>(
        x_bf, wk_bf, k_raw, NTOK, KVOUT, HID);
    gemm_bf16_wmma<0><<<dim3(KVOUT / 64, NTOK / 128), 256, 0, stream>>>(
        x_bf, wv_bf, v_raw, NTOK, KVOUT, HID);
    gemm_bf16_wmma<0><<<dim3(KVOUT / 64, NCTX / 128), 256, 0, stream>>>(
        hctx_bf, wkc_bf, kc_raw, NCTX, KVOUT, HID);
    gemm_bf16_wmma<0><<<dim3(KVOUT / 64, NCTX / 128), 256, 0, stream>>>(
        hctx_bf, wvc_bf, vc_raw, NCTX, KVOUT, HID);

    // ---- 3) RMS-norm (+RoPE) and KV-cache assembly
    post_norm_rope<<<BATCH * SEQ_L * NHEADS, 128, 0, stream>>>(
        q_raw, q_att, qnw, SEQ_L, NHEADS, SEQ_L, 0, 1, 0, SCALE_QK); // q: norm+rope+scale
    post_norm_rope<<<BATCH * SEQ_L * NKVH, 128, 0, stream>>>(
        k_raw, k_full, knw, SEQ_L, NKVH, SEQ_S, SEQ_T, 1, 1, 1.0f);  // k: norm+rope, d-major
    post_norm_rope<<<BATCH * SEQ_T * NKVH, 128, 0, stream>>>(
        kc_raw, k_full, knw, SEQ_T, NKVH, SEQ_S, 0, 0, 1, 1.0f);     // k_ctx: norm, d-major
    scatter_v<<<2048, 256, 0, stream>>>(v_raw, v_full, SEQ_L, SEQ_T,
                                        (size_t)NTOK * KVOUT);
    scatter_v<<<2048, 256, 0, stream>>>(vc_raw, v_full, SEQ_T, 0,
                                        (size_t)NCTX * KVOUT);

    // ---- 4) flash attention
    attn_wmma<<<dim3(SEQ_L / 64, BATCH * NHEADS), 128, 0, stream>>>(
        q_att, k_full, v_full, attn_bf);

    // ---- 5) output projection -> f32 d_out
    gemm_bf16_wmma<1><<<dim3(HID / 64, NTOK / 128), 256, 0, stream>>>(
        attn_bf, wo_bf, out, NTOK, HID, QOUT);
}
